// TransformerBlock_34119220199745
// MI455X (gfx1250) — matrix operand order repaired and checked by reading, not compiled, not run on hardware
//
#include <hip/hip_runtime.h>
#include <hip/hip_bf16.h>
#include <math.h>
#include <stdint.h>

// Problem constants (match reference)
constexpr int BB = 2;
constexpr int TT = 2048;
constexpr int CC = 1024;
constexpr int HH = 16;
constexpr int DH = 64;
constexpr int FFN = 4096;
constexpr int MM = BB * TT;   // 4096 rows total

typedef __bf16 bf16;
typedef __bf16 v16bf __attribute__((ext_vector_type(16)));
typedef __bf16 v8bf  __attribute__((ext_vector_type(8)));
typedef float  v8f   __attribute__((ext_vector_type(8)));

static __device__ __forceinline__ v16bf cat16(v8bf a, v8bf b) {
  return __builtin_shufflevector(a, b, 0,1,2,3,4,5,6,7,8,9,10,11,12,13,14,15);
}
static __device__ __forceinline__ v8f wmma_bf16(v16bf a, v16bf b, v8f c) {
  // D = A(16x32 bf16) * B(32x16 bf16) + C(16x16 f32)
  return __builtin_amdgcn_wmma_f32_16x16x32_bf16(false, a, false, b, (short)0, c, false, false);
}

// Async DMA of 16 bytes global -> LDS (ASYNCcnt-tracked), GV addressing mode.
static __device__ __forceinline__ void async_copy_b128(const bf16* g, bf16* l) {
  unsigned loff = (unsigned)(uintptr_t)l;          // LDS aperture: addr[31:0] = LDS offset
  uint64_t gaddr = (uint64_t)(uintptr_t)g;
  asm volatile("global_load_async_to_lds_b128 %0, %1, off"
               :: "v"(loff), "v"(gaddr) : "memory");
}
static __device__ __forceinline__ void wait_async0() {
  asm volatile("s_wait_asynccnt 0x0" ::: "memory");
}

// ---------------- fp32 -> bf16 conversion (weights) ----------------
__global__ __launch_bounds__(256) void cvt_kernel(const float* __restrict__ in,
                                                  bf16* __restrict__ out, int n) {
  for (int i = blockIdx.x * blockDim.x + threadIdx.x; i < n; i += gridDim.x * blockDim.x)
    out[i] = (bf16)in[i];
}

// ---------------- LayerNorm: fp32 row -> bf16 row ----------------
__global__ __launch_bounds__(256) void ln_kernel(const float* __restrict__ x,
                                                 const float* __restrict__ w,
                                                 const float* __restrict__ bvec,
                                                 bf16* __restrict__ out) {
  __shared__ float red[8];
  int row = blockIdx.x;
  int tid = threadIdx.x;
  const float4* xr = (const float4*)(x + (size_t)row * CC);
  float4 v = xr[tid];                       // 256 threads * 4 = 1024 = CC
  float s = v.x + v.y + v.z + v.w;
#pragma unroll
  for (int off = 16; off > 0; off >>= 1) s += __shfl_xor(s, off, 32);
  if ((tid & 31) == 0) red[tid >> 5] = s;
  __syncthreads();
  float tot = 0.0f;
#pragma unroll
  for (int i = 0; i < 8; i++) tot += red[i];
  float mu = tot * (1.0f / CC);
  __syncthreads();
  float d0 = v.x - mu, d1 = v.y - mu, d2 = v.z - mu, d3 = v.w - mu;
  float sq = d0 * d0 + d1 * d1 + d2 * d2 + d3 * d3;
#pragma unroll
  for (int off = 16; off > 0; off >>= 1) sq += __shfl_xor(sq, off, 32);
  if ((tid & 31) == 0) red[tid >> 5] = sq;
  __syncthreads();
  float vt = 0.0f;
#pragma unroll
  for (int i = 0; i < 8; i++) vt += red[i];
  float rstd = rsqrtf(vt * (1.0f / CC) + 1e-5f);
  int c = tid * 4;
  float4 wv = *(const float4*)(w + c);
  float4 bv = *(const float4*)(bvec + c);
  bf16* o = out + (size_t)row * CC + c;
  o[0] = (bf16)(d0 * rstd * wv.x + bv.x);
  o[1] = (bf16)(d1 * rstd * wv.y + bv.y);
  o[2] = (bf16)(d2 * rstd * wv.z + bv.z);
  o[3] = (bf16)(d3 * rstd * wv.w + bv.w);
}

// ---------------- WMMA GEMM: D[M,N] = A[M,K](bf16) * W[N,K]^T(bf16) ----------------
// Wave tile 32x64 (8 accumulators). B tile (64x32) staged in LDS via async DMA,
// double buffered, shared by the 8 waves of the block (block tile 256x64).
// MODE 0: store bf16 ; MODE 1: exact GELU -> bf16 ; MODE 2: + resid -> f32
template <int MODE>
__global__ __launch_bounds__(256) void gemm_kernel(const bf16* __restrict__ A,
                                                   const bf16* __restrict__ W,
                                                   int N, int K,
                                                   const float* __restrict__ resid,
                                                   float* __restrict__ outF,
                                                   bf16* __restrict__ outB) {
  constexpr int PITCH = 40;   // bf16 elems per LDS row (80 B) -> conflict-free b128 reads
  __shared__ __align__(16) bf16 btile[2][64 * PITCH];

  int tid = threadIdx.x;
  int lane = tid & 31, wave = tid >> 5;
  int hi = lane >> 4, lo = lane & 15;
  int m0 = blockIdx.y * 256 + wave * 32;  // 8 waves x 32 rows
  int n0 = blockIdx.x * 64;

  // Async staging role: 256 threads x 16B = one 64x32 bf16 tile.
  int srow = tid >> 2;                    // 0..63  (tile row = n within block N-tile)
  int sq   = tid & 3;                     // 0..3   (16B chunk within 64B row)
  const bf16* gsrc = W + (size_t)(n0 + srow) * K + sq * 8;
  bf16* ldst0 = &btile[0][srow * PITCH + sq * 8];
  bf16* ldst1 = &btile[1][srow * PITCH + sq * 8];

  v8f acc[2][4];
#pragma unroll
  for (int mh = 0; mh < 2; mh++)
#pragma unroll
    for (int i = 0; i < 4; i++)
#pragma unroll
      for (int j = 0; j < 8; j++) acc[mh][i][j] = 0.0f;

  // A fragments: lane -> rows m0+lo and m0+16+lo, K chunks [kb,kb+8) & [kb+16,kb+24)
  const bf16* arow0 = A + (size_t)(m0 + lo) * K + hi * 8;
  const bf16* arow1 = arow0 + (size_t)16 * K;

  int nsteps = K >> 5;
  async_copy_b128(gsrc, ldst0);           // prologue: stage k-step 0 into buf0
  for (int s = 0; s < nsteps; ++s) {
    int kk = s << 5;
    wait_async0();                        // my chunk of current tile landed
    __syncthreads();                      // everyone's chunks landed / prior readers done
    if (s + 1 < nsteps)                   // stage next tile into the other buffer
      async_copy_b128(gsrc + kk + 32, (s & 1) ? ldst0 : ldst1);

    const bf16* bt = &btile[s & 1][0];
    v16bf a0 = cat16(*(const v8bf*)(arow0 + kk), *(const v8bf*)(arow0 + kk + 16));
    v16bf a1 = cat16(*(const v8bf*)(arow1 + kk), *(const v8bf*)(arow1 + kk + 16));
#pragma unroll
    for (int nt = 0; nt < 4; nt++) {
      const bf16* bp = bt + (nt * 16 + lo) * PITCH + hi * 8;
      v16bf bfr = cat16(*(const v8bf*)bp, *(const v8bf*)(bp + 16));
      acc[0][nt] = wmma_bf16(a0, bfr, acc[0][nt]);
      acc[1][nt] = wmma_bf16(a1, bfr, acc[1][nt]);
    }
  }

#pragma unroll
  for (int mh = 0; mh < 2; mh++)
#pragma unroll
    for (int nt = 0; nt < 4; nt++)
#pragma unroll
      for (int r = 0; r < 8; r++) {
        int row = m0 + mh * 16 + r + 8 * hi;   // D layout: M = r + 8*hi, N = lo
        int col = n0 + nt * 16 + lo;
        size_t idx = (size_t)row * N + col;
        float v = acc[mh][nt][r];
        if (MODE == 0) {
          outB[idx] = (bf16)v;
        } else if (MODE == 1) {
          v = 0.5f * v * (1.0f + erff(v * 0.70710678118654752f));
          outB[idx] = (bf16)v;
        } else {
          outF[idx] = v + resid[idx];
        }
      }
}

// ---------------- V transpose: qkv[b,t, 2C + h*Dh + d] -> vt[b,h,d,t] ----------------
__global__ __launch_bounds__(256) void vtrans_kernel(const bf16* __restrict__ qkv,
                                                     bf16* __restrict__ vt) {
  int i = blockIdx.x * blockDim.x + threadIdx.x;  // grid sized exactly
  int t = i & (TT - 1);
  int rest = i >> 11;          // TT = 2^11
  int d = rest & (DH - 1);
  rest >>= 6;                  // DH = 2^6
  int h = rest & (HH - 1);
  int b = rest >> 4;           // HH = 2^4
  vt[i] = qkv[(size_t)(b * TT + t) * (3 * CC) + 2 * CC + h * DH + d];
}

// ---------------- Flash attention: one wave per 16-row query tile ----------------
constexpr int ATT_WAVES = 4;
__global__ __launch_bounds__(128) void attn_kernel(const bf16* __restrict__ qkv,
                                                   const bf16* __restrict__ vt,
                                                   bf16* __restrict__ attn) {
  __shared__ __align__(16) bf16 plds[ATT_WAVES][16 * 32];
  int lane = threadIdx.x & 31;
  int wave = threadIdx.x >> 5;
  int hi = lane >> 4, lo = lane & 15;
  int qtile = blockIdx.x * ATT_WAVES + wave;   // 0..127
  int bh = blockIdx.y;                         // 0..31
  int b = bh >> 4, h = bh & 15;
  int q0 = qtile * 16;

  // Q fragments for d-windows [0,32) and [32,64)
  const bf16* qbase = qkv + (size_t)(b * TT + q0 + lo) * (3 * CC) + h * DH + hi * 8;
  v16bf qf0 = cat16(*(const v8bf*)(qbase),      *(const v8bf*)(qbase + 16));
  v16bf qf1 = cat16(*(const v8bf*)(qbase + 32), *(const v8bf*)(qbase + 48));

  v8f o[4];
  float mrun[8], lrun[8];
#pragma unroll
  for (int i = 0; i < 4; i++)
#pragma unroll
    for (int j = 0; j < 8; j++) o[i][j] = 0.0f;
#pragma unroll
  for (int r = 0; r < 8; r++) { mrun[r] = -INFINITY; lrun[r] = 0.0f; }

  const bf16* kbase = qkv + (size_t)b * TT * (3 * CC) + CC + h * DH;
  const bf16* vbase = vt + (size_t)bh * DH * TT;
  bf16* pl = &plds[wave][0];

  int kvlimit = q0 + 15;                 // causal: kv <= query row
  for (int kv0 = 0; kv0 <= kvlimit; kv0 += 32) {
    // ---- scores S = Q K^T over 32 kv columns (2 N-tiles x 2 K-steps) ----
    v8f s[2];
#pragma unroll
    for (int nt = 0; nt < 2; nt++)
#pragma unroll
      for (int j = 0; j < 8; j++) s[nt][j] = 0.0f;
#pragma unroll
    for (int nt = 0; nt < 2; nt++) {
      const bf16* kp = kbase + (size_t)(kv0 + nt * 16 + lo) * (3 * CC) + hi * 8;
      v16bf kb0 = cat16(*(const v8bf*)(kp),      *(const v8bf*)(kp + 16));
      v16bf kb1 = cat16(*(const v8bf*)(kp + 32), *(const v8bf*)(kp + 48));
      s[nt] = wmma_bf16(qf0, kb0, s[nt]);
      s[nt] = wmma_bf16(qf1, kb1, s[nt]);
    }
    // ---- scale + causal mask + row max (rows live across a 16-lane half) ----
    float rowmax[8], scale[8], rowsum[8];
#pragma unroll
    for (int r = 0; r < 8; r++) {
      int qt = q0 + r + 8 * hi;
#pragma unroll
      for (int nt = 0; nt < 2; nt++) {
        int kv = kv0 + nt * 16 + lo;
        float sv = s[nt][r] * 0.125f;    // 1/sqrt(Dh)
        s[nt][r] = (kv <= qt) ? sv : -INFINITY;
      }
      rowmax[r] = fmaxf(s[0][r], s[1][r]);
    }
#pragma unroll
    for (int r = 0; r < 8; r++)
      for (int off = 1; off < 16; off <<= 1)
        rowmax[r] = fmaxf(rowmax[r], __shfl_xor(rowmax[r], off, 32));
    // ---- online softmax update ----
#pragma unroll
    for (int r = 0; r < 8; r++) {
      float mnew = fmaxf(mrun[r], rowmax[r]);
      scale[r] = __expf(mrun[r] - mnew);
      mrun[r] = mnew;
      rowsum[r] = 0.0f;
#pragma unroll
      for (int nt = 0; nt < 2; nt++) {
        float pe = __expf(s[nt][r] - mnew);
        s[nt][r] = pe;
        rowsum[r] += pe;
      }
    }
#pragma unroll
    for (int r = 0; r < 8; r++)
      for (int off = 1; off < 16; off <<= 1)
        rowsum[r] += __shfl_xor(rowsum[r], off, 32);
#pragma unroll
    for (int r = 0; r < 8; r++) lrun[r] = lrun[r] * scale[r] + rowsum[r];
#pragma unroll
    for (int nt = 0; nt < 4; nt++)
#pragma unroll
      for (int r = 0; r < 8; r++) o[nt][r] *= scale[r];
    // ---- stage P (D-layout) -> LDS -> A-fragment layout ----
#pragma unroll
    for (int nt = 0; nt < 2; nt++)
#pragma unroll
      for (int r = 0; r < 8; r++)
        pl[(r + 8 * hi) * 32 + nt * 16 + lo] = (bf16)s[nt][r];
    __asm__ volatile("s_wait_dscnt 0" ::: "memory");
    v16bf pf = cat16(*(const v8bf*)(pl + lo * 32 + hi * 8),
                     *(const v8bf*)(pl + lo * 32 + hi * 8 + 16));
    // ---- O += P * V (4 N-tiles over Dh, one K-step of 32 kv) ----
#pragma unroll
    for (int nt = 0; nt < 4; nt++) {
      const bf16* vp = vbase + (size_t)(nt * 16 + lo) * TT + kv0 + hi * 8;
      v16bf vf = cat16(*(const v8bf*)vp, *(const v8bf*)(vp + 16));
      o[nt] = wmma_bf16(pf, vf, o[nt]);
    }
  }
  // ---- normalize and store bf16 [B,T,C] ----
#pragma unroll
  for (int nt = 0; nt < 4; nt++)
#pragma unroll
    for (int r = 0; r < 8; r++) {
      size_t idx = (size_t)(b * TT + q0 + r + 8 * hi) * CC + h * DH + nt * 16 + lo;
      attn[idx] = (bf16)(o[nt][r] / lrun[r]);
    }
}

// ---------------- host launcher ----------------
extern "C" void kernel_launch(void* const* d_in, const int* in_sizes, int n_in,
                              void* d_out, int out_size, void* d_ws, size_t ws_size,
                              hipStream_t stream) {
  (void)in_sizes; (void)n_in; (void)out_size; (void)ws_size;
  const float* x    = (const float*)d_in[0];
  const float* ln1w = (const float*)d_in[1];
  const float* ln1b = (const float*)d_in[2];
  const float* qkvw = (const float*)d_in[3];
  const float* outw = (const float*)d_in[4];
  const float* ln2w = (const float*)d_in[5];
  const float* ln2b = (const float*)d_in[6];
  const float* w1   = (const float*)d_in[7];
  const float* w2   = (const float*)d_in[8];
  float* out = (float*)d_out;

  char* p = (char*)d_ws;
  auto take = [&](size_t bytes) { char* r = p; p += (bytes + 255) & ~(size_t)255; return r; };
  bf16*  h_b    = (bf16*)take((size_t)MM * CC * 2);          // ln1 out, reused for ln2 out
  bf16*  qkv_b  = (bf16*)take((size_t)MM * 3 * CC * 2);
  bf16*  vt_b   = (bf16*)take((size_t)BB * HH * DH * TT * 2);
  bf16*  attn_b = (bf16*)take((size_t)MM * CC * 2);
  float* x1     = (float*)take((size_t)MM * CC * 4);
  bf16*  hid_b  = (bf16*)take((size_t)MM * FFN * 2);
  bf16*  wq_b   = (bf16*)take((size_t)3 * CC * CC * 2);
  bf16*  wo_b   = (bf16*)take((size_t)CC * CC * 2);
  bf16*  w1_b   = (bf16*)take((size_t)FFN * CC * 2);
  bf16*  w2_b   = (bf16*)take((size_t)CC * FFN * 2);

  // 1) weights -> bf16 (L2-resident thereafter)
  cvt_kernel<<<2048, 256, 0, stream>>>(qkvw, wq_b, 3 * CC * CC);
  cvt_kernel<<<2048, 256, 0, stream>>>(outw, wo_b, CC * CC);
  cvt_kernel<<<2048, 256, 0, stream>>>(w1, w1_b, FFN * CC);
  cvt_kernel<<<2048, 256, 0, stream>>>(w2, w2_b, CC * FFN);

  // 2) pre-LN 1
  ln_kernel<<<MM, 256, 0, stream>>>(x, ln1w, ln1b, h_b);
  // 3) QKV projection: [4096,1024] x [3072,1024]^T -> bf16 [4096,3072]
  gemm_kernel<0><<<dim3(3 * CC / 64, MM / 256), 256, 0, stream>>>(
      h_b, wq_b, 3 * CC, CC, nullptr, nullptr, qkv_b);
  // 4) V -> [b,h,d,t] for contiguous P*V B-fragments
  vtrans_kernel<<<(BB * HH * DH * TT) / 256, 256, 0, stream>>>(qkv_b, vt_b);
  // 5) causal flash attention
  attn_kernel<<<dim3(TT / 16 / ATT_WAVES, BB * HH), 32 * ATT_WAVES, 0, stream>>>(
      qkv_b, vt_b, attn_b);
  // 6) output projection + residual -> x1 (f32)
  gemm_kernel<2><<<dim3(CC / 64, MM / 256), 256, 0, stream>>>(
      attn_b, wo_b, CC, CC, x, x1, nullptr);
  // 7) pre-LN 2
  ln_kernel<<<MM, 256, 0, stream>>>(x1, ln2w, ln2b, h_b);
  // 8) FFN up + exact GELU -> bf16 [4096,4096]
  gemm_kernel<1><<<dim3(FFN / 64, MM / 256), 256, 0, stream>>>(
      h_b, w1_b, FFN, CC, nullptr, nullptr, hid_b);
  // 9) FFN down + residual -> final f32 output
  gemm_kernel<2><<<dim3(CC / 64, MM / 256), 256, 0, stream>>>(
      hid_b, w2_b, CC, FFN, x1, out, nullptr);
}
